// SSEmbGNN_75823352644057
// MI455X (gfx1250) — compile-verified, matrix operand order checked by reading, not run on hardware
//
#include <hip/hip_runtime.h>

// ---------------------------------------------------------------------------
// Types
// ---------------------------------------------------------------------------
typedef __bf16 bf16_t;
typedef __attribute__((ext_vector_type(16))) __bf16 v16bf;
typedef __attribute__((ext_vector_type(8)))  float   v8f;
typedef __attribute__((ext_vector_type(4)))  unsigned int u32x4;
typedef __attribute__((ext_vector_type(8)))  int i32x8;
typedef __attribute__((ext_vector_type(4)))  int i32x4;

#define NS   256
#define NV   64
#define ES   64
#define EV   16
#define NAA  21
#define MSAD 768

__device__ inline bf16_t f2b(float x) {
    unsigned u = __float_as_uint(x);
    u += 0x7FFFu + ((u >> 16) & 1u);
    unsigned short h = (unsigned short)(u >> 16);
    bf16_t r; __builtin_memcpy(&r, &h, 2); return r;
}
__device__ inline float b2f(bf16_t x) {
    unsigned short h; __builtin_memcpy(&h, &x, 2);
    return __uint_as_float(((unsigned)h) << 16);
}
__device__ inline bf16_t to_bfv(float x)  { return f2b(x); }
__device__ inline bf16_t to_bfv(bf16_t x) { return x; }

// ---------------------------------------------------------------------------
// Tensor Data Mover (async tile load global->LDS), CDNA5. Probe-verified
// builtin arity: ROCm7.2/clang-22 = 5 args, clang-23 toolchain = 6 args.
// ---------------------------------------------------------------------------
#if __has_builtin(__builtin_amdgcn_tensor_load_to_lds)
#define HAVE_TDM 1
__device__ inline void tdm_call(u32x4 g0, i32x8 g1) {
    i32x4 z4 = {0, 0, 0, 0};
#if __clang_major__ >= 23
    i32x8 z8 = {0, 0, 0, 0, 0, 0, 0, 0};
    __builtin_amdgcn_tensor_load_to_lds(g0, g1, z4, z4, z8, 0);
#else
    __builtin_amdgcn_tensor_load_to_lds(g0, g1, z4, z4, 0);
#endif
}
#else
#define HAVE_TDM 0
#endif

// ---------------------------------------------------------------------------
// WMMA bf16 GEMM:  C[M,N] (f32) = A[M,K](bf16) @ W[N,K](bf16)^T  (+bias)
// Block: 128x64 tile, 8 wave32 waves; each wave owns a 32x32 patch =
// 2x2 tiles of V_WMMA_F32_16X16X32_BF16 (4 WMMAs per 4 fragment loads).
// Staging: TDM tensor_load_to_lds (TENSORcnt) with hardware pad to the
// 80B LDS row, else per-thread b128 double-buffering fallback.
// ---------------------------------------------------------------------------
#define BM  128
#define BN  64
#define BKK 32
#define LDK 40   // padded LDS row (bf16) -> 80B stride

__device__ inline u32x4 ld16g(const bf16_t* base, long row, int nrows, int k0, int K) {
    u32x4 r = {0u, 0u, 0u, 0u};
    if (row < (long)nrows) {
        const bf16_t* p = base + row * (long)K + k0;
        if (k0 + 8 <= K && ((((unsigned long long)(size_t)p) & 15ull) == 0ull)) {
            r = *(const u32x4*)p;
        } else {
            unsigned short t[8];
#pragma unroll
            for (int i = 0; i < 8; ++i) {
                unsigned short h = 0;
                if (k0 + i < K) __builtin_memcpy(&h, p + i, 2);
                t[i] = h;
            }
            r.x = (unsigned)t[0] | ((unsigned)t[1] << 16);
            r.y = (unsigned)t[2] | ((unsigned)t[3] << 16);
            r.z = (unsigned)t[4] | ((unsigned)t[5] << 16);
            r.w = (unsigned)t[6] | ((unsigned)t[7] << 16);
        }
    }
    return r;
}

__device__ inline v16bf frag_ld(const bf16_t* p) {
    union { v16bf v; u32x4 q[2]; } u;
    u.q[0] = *(const u32x4*)(p);        // k = half*8 .. +7   (VGPR 0-3)
    u.q[1] = *(const u32x4*)(p + 16);   // k = 16+half*8 ..   (VGPR 4-7)
    return u.v;
}

#if HAVE_TDM
// Issue one TDM descriptor moving a [tile_rows x 32] bf16 tile starting at
// (row0, k0) of a row-major [nrows x K] matrix into LDS at lds_off, padding
// each 64B row to 80B (pad_interval=16 DWORDs, pad_amount=4 DWORDs).
__device__ inline void tdm_tile(const bf16_t* g, long row0, int nrows,
                                int K, int k0, int tile_rows, unsigned lds_off)
{
    unsigned long long ga = (unsigned long long)(size_t)(g + row0 * (long)K + k0);
    unsigned d0 = (unsigned)(K - k0 > 0 ? K - k0 : 0);          // remaining K elems
    long rr = (long)nrows - row0;
    unsigned d1 = (unsigned)(rr > 0 ? rr : 0);                  // remaining rows
    u32x4 g0;
    g0.x = 1u;                                                  // count=1 user D#
    g0.y = lds_off;                                             // lds_addr
    g0.z = (unsigned)(ga & 0xFFFFFFFFull);                      // global_addr lo
    g0.w = (unsigned)((ga >> 32) & 0x01FFFFFFull) | (2u << 30); // addr hi | type=2
    i32x8 g1;
    g1[0] = (int)((1u << 16) |                 // data_size = 2 bytes
                  (1u << 20) |                 // pad_enable
                  (3u << 22) |                 // pad_interval: 16 DWORDs
                  (3u << 25));                 // pad_amount: 4 DWORDs (16B)
    g1[1] = (int)((d0 & 0xFFFFu) << 16);                        // tensor_dim0 lo16
    g1[2] = (int)((d0 >> 16) | ((d1 & 0xFFFFu) << 16));         // dim0 hi | dim1 lo
    g1[3] = (int)((d1 >> 16) | ((unsigned)BKK << 16));          // dim1 hi | tile_dim0
    g1[4] = (int)(unsigned)tile_rows;                           // tile_dim1 (tile_dim2=0)
    g1[5] = (int)(unsigned)K;                                   // tensor_dim0_stride lo32
    g1[6] = 0;
    g1[7] = 0;
    tdm_call(g0, g1);
}
#endif

__global__ __launch_bounds__(256)
void k_gemm(const bf16_t* __restrict__ A, const bf16_t* __restrict__ W,
            const float* __restrict__ bias,
            float* __restrict__ C, bf16_t* __restrict__ Cb,
            int M, int N, int K)
{
    __shared__ bf16_t sA[2][BM][LDK];
    __shared__ bf16_t sW[2][BN][LDK];

    const int t    = threadIdx.x;
    const int lane = t & 31;
    const int wave = t >> 5;
    const int bm   = blockIdx.x * BM;
    const int bn   = blockIdx.y * BN;
    const int wm   = (wave & 3) * 32;    // wave 32-row patch
    const int wn   = (wave >> 2) * 32;   // wave 32-col patch
    const int fr   = lane & 15;
    const int fh   = lane >> 4;

    v8f acc00 = {}, acc01 = {}, acc10 = {}, acc11 = {};
    const int nk = (K + BKK - 1) / BKK;

#if HAVE_TDM
    unsigned ldsA[2] = { (unsigned)(size_t)(void*)&sA[0][0][0],
                         (unsigned)(size_t)(void*)&sA[1][0][0] };
    unsigned ldsW[2] = { (unsigned)(size_t)(void*)&sW[0][0][0],
                         (unsigned)(size_t)(void*)&sW[1][0][0] };
    if (wave == 0) {
        tdm_tile(A, bm, M, K, 0, BM, ldsA[0]);
        tdm_tile(W, bn, N, K, 0, BN, ldsW[0]);
        __builtin_amdgcn_s_wait_tensorcnt((short)0);
    }
    __syncthreads();
    for (int kt = 0; kt < nk; ++kt) {
        const int cur = kt & 1;
        if (wave == 0 && kt + 1 < nk) {
            tdm_tile(A, bm, M, K, (kt + 1) * BKK, BM, ldsA[cur ^ 1]);
            tdm_tile(W, bn, N, K, (kt + 1) * BKK, BN, ldsW[cur ^ 1]);
        }
        v16bf a0 = frag_ld(&sA[cur][wm + fr][fh * 8]);
        v16bf a1 = frag_ld(&sA[cur][wm + 16 + fr][fh * 8]);
        v16bf b0 = frag_ld(&sW[cur][wn + fr][fh * 8]);
        v16bf b1 = frag_ld(&sW[cur][wn + 16 + fr][fh * 8]);
        acc00 = __builtin_amdgcn_wmma_f32_16x16x32_bf16(false, a0, false, b0, (short)0, acc00, false, false);
        acc01 = __builtin_amdgcn_wmma_f32_16x16x32_bf16(false, a0, false, b1, (short)0, acc01, false, false);
        acc10 = __builtin_amdgcn_wmma_f32_16x16x32_bf16(false, a1, false, b0, (short)0, acc10, false, false);
        acc11 = __builtin_amdgcn_wmma_f32_16x16x32_bf16(false, a1, false, b1, (short)0, acc11, false, false);
        if (wave == 0 && kt + 1 < nk) __builtin_amdgcn_s_wait_tensorcnt((short)0);
        __syncthreads();
    }
#else
    // per-thread b128 staging, register double buffer
    const int aR0 = t >> 2, aC = (t & 3) * 8;   // A: 2 chunks/thread (512 total)
    const int aR1 = aR0 + 64;
    u32x4 ra0 = ld16g(A, (long)(bm + aR0), M, aC, K);
    u32x4 ra1 = ld16g(A, (long)(bm + aR1), M, aC, K);
    u32x4 rw  = ld16g(W, (long)(bn + aR0), N, aC, K);
    *(u32x4*)&sA[0][aR0][aC] = ra0;
    *(u32x4*)&sA[0][aR1][aC] = ra1;
    *(u32x4*)&sW[0][aR0][aC] = rw;
    for (int kt = 0; kt < nk; ++kt) {
        __syncthreads();
        const int cur = kt & 1;
        if (kt + 1 < nk) {
            const int k0 = (kt + 1) * BKK + aC;
            ra0 = ld16g(A, (long)(bm + aR0), M, k0, K);
            ra1 = ld16g(A, (long)(bm + aR1), M, k0, K);
            rw  = ld16g(W, (long)(bn + aR0), N, k0, K);
        }
        if (kt + 2 < nk) {  // global_prefetch_b8 two tiles ahead
            if (bm + aR0 < M) __builtin_prefetch(A + (long)(bm + aR0) * K + (kt + 2) * BKK + aC, 0, 1);
            if (bn + aR0 < N) __builtin_prefetch(W + (long)(bn + aR0) * K + (kt + 2) * BKK + aC, 0, 1);
        }
        v16bf a0 = frag_ld(&sA[cur][wm + fr][fh * 8]);
        v16bf a1 = frag_ld(&sA[cur][wm + 16 + fr][fh * 8]);
        v16bf b0 = frag_ld(&sW[cur][wn + fr][fh * 8]);
        v16bf b1 = frag_ld(&sW[cur][wn + 16 + fr][fh * 8]);
        acc00 = __builtin_amdgcn_wmma_f32_16x16x32_bf16(false, a0, false, b0, (short)0, acc00, false, false);
        acc01 = __builtin_amdgcn_wmma_f32_16x16x32_bf16(false, a0, false, b1, (short)0, acc01, false, false);
        acc10 = __builtin_amdgcn_wmma_f32_16x16x32_bf16(false, a1, false, b0, (short)0, acc10, false, false);
        acc11 = __builtin_amdgcn_wmma_f32_16x16x32_bf16(false, a1, false, b1, (short)0, acc11, false, false);
        if (kt + 1 < nk) {
            const int nx = cur ^ 1;
            *(u32x4*)&sA[nx][aR0][aC] = ra0;
            *(u32x4*)&sA[nx][aR1][aC] = ra1;
            *(u32x4*)&sW[nx][aR0][aC] = rw;
        }
    }
#endif

    const int c0  = bn + wn + fr;
    const int c1  = c0 + 16;
    const float bb0 = (bias && c0 < N) ? bias[c0] : 0.f;
    const float bb1 = (bias && c1 < N) ? bias[c1] : 0.f;
#pragma unroll
    for (int g = 0; g < 8; ++g) {
        const int r0 = bm + wm + fh * 8 + g;
        const int r1 = r0 + 16;
        if (r0 < M) {
            if (c0 < N) { float x = acc00[g] + bb0; C[(long)r0 * N + c0] = x; if (Cb) Cb[(long)r0 * N + c0] = f2b(x); }
            if (c1 < N) { float x = acc01[g] + bb1; C[(long)r0 * N + c1] = x; if (Cb) Cb[(long)r0 * N + c1] = f2b(x); }
        }
        if (r1 < M) {
            if (c0 < N) { float x = acc10[g] + bb0; C[(long)r1 * N + c0] = x; if (Cb) Cb[(long)r1 * N + c0] = f2b(x); }
            if (c1 < N) { float x = acc11[g] + bb1; C[(long)r1 * N + c1] = x; if (Cb) Cb[(long)r1 * N + c1] = f2b(x); }
        }
    }
}

// ---------------------------------------------------------------------------
// Elementwise / reduction kernels
// ---------------------------------------------------------------------------
__global__ void k_map(const float* x, bf16_t* y, long n, int mode) {
    long i = (long)blockIdx.x * 256 + threadIdx.x;
    if (i >= n) return;
    float v = x[i];
    if (mode == 1) v = fmaxf(v, 0.f);
    else if (mode == 2) v = 1.f / (1.f + __expf(-v));
    y[i] = f2b(v);
}
__global__ void k_relu_ip(float* x, long n) {
    long i = (long)blockIdx.x * 256 + threadIdx.x;
    if (i < n) x[i] = fmaxf(x[i], 0.f);
}
__global__ void k_add_ip(float* a, const float* b, long n) {
    long i = (long)blockIdx.x * 256 + threadIdx.x;
    if (i < n) a[i] += b[i];
}
__global__ void k_copy_f(float* d, const float* s, long n) {
    long i = (long)blockIdx.x * 256 + threadIdx.x;
    if (i < n) d[i] = s[i];
}
// vn[r,h] = sqrt(max(sum_c vh[r,c,h]^2, eps)), vh layout [(r*3+c)*H + h]
__global__ void k_vnorm(const float* vh, float* vn, long R, int H) {
    long tot = R * (long)H;
    long i = (long)blockIdx.x * 256 + threadIdx.x;
    if (i >= tot) return;
    long r = i / H; int h = (int)(i - r * H);
    const float* p = vh + (r * 3) * (long)H + h;
    float a = p[0], b = p[H], c = p[2L * H];
    vn[i] = sqrtf(fmaxf(a * a + b * b + c * c, 1e-8f));
}
template <typename TA, typename TB>
__global__ void k_cat(const TA* a, int da, const TB* b, int db, bf16_t* o, long R) {
    long D = da + db;
    long i = (long)blockIdx.x * 256 + threadIdx.x;
    if (i >= R * D) return;
    long r = i / D; int c = (int)(i - r * D);
    o[i] = (c < da) ? to_bfv(a[r * da + c]) : to_bfv(b[r * db + (c - da)]);
}
// v[r,c,o] *= sigmoid(gate[r,o]); optional bf16 mirror
__global__ void k_gate_mul(float* v, bf16_t* vb, const float* gate, long R, int VO) {
    long tot = R * 3L * VO;
    long i = (long)blockIdx.x * 256 + threadIdx.x;
    if (i >= tot) return;
    long r = i / (3L * VO);
    int o = (int)(i % VO);
    float g = 1.f / (1.f + __expf(-gate[r * VO + o]));
    float x = v[i] * g;
    v[i] = x;
    if (vb) vb[i] = f2b(x);
}
// [R,V,3] f32 -> [R,3,V] bf16
__global__ void k_tr_v(const float* in, bf16_t* out, long R, int V) {
    long tot = R * 3L * V;
    long i = (long)blockIdx.x * 256 + threadIdx.x;
    if (i >= tot) return;
    long r = i / (3L * V);
    int rem = (int)(i - r * 3L * V);
    int c = rem / V, j = rem % V;
    out[i] = f2b(in[(r * V + j) * 3 + c]);
}
__global__ void k_gather_hs(const float* WS, const int* seq, const int* src,
                            bf16_t* out, long E, int naa) {
    long tot = E * (long)naa;
    long i = (long)blockIdx.x * 256 + threadIdx.x;
    if (i >= tot) return;
    long e = i / naa; int j = (int)(i % naa);
    out[i] = f2b(WS[(long)seq[src[e]] * naa + j]);
}
__global__ void k_count(const int* dst, float* cnt, long E) {
    long e = (long)blockIdx.x * 256 + threadIdx.x;
    if (e < E) atomicAdd(&cnt[dst[e]], 1.f);
}
__global__ void k_scatter_add(const float* x, const int* dst, float* acc, long E, int D) {
    long tot = E * (long)D;
    long i = (long)blockIdx.x * 256 + threadIdx.x;
    if (i >= tot) return;
    long e = i / D; int d = (int)(i % D);
    atomicAdd(&acc[(long)dst[e] * D + d], x[i]);
}
__global__ void k_mean_res(float* s, const float* acc, const float* cnt, long Nn, int D) {
    long tot = Nn * (long)D;
    long i = (long)blockIdx.x * 256 + threadIdx.x;
    if (i >= tot) return;
    long n = i / D;
    s[i] += acc[i] / fmaxf(cnt[n], 1.f);
}
// row layernorm in place (block per row, LDS tree reduce)
__global__ void k_ln(float* x, const float* w, const float* b, int D) {
    __shared__ float red[256];
    float* row = x + (long)blockIdx.x * D;
    int t = threadIdx.x;
    float s = 0.f;
    for (int i = t; i < D; i += 256) s += row[i];
    red[t] = s; __syncthreads();
    for (int k = 128; k > 0; k >>= 1) { if (t < k) red[t] += red[t + k]; __syncthreads(); }
    float mu = red[0] / D;
    __syncthreads();
    float vv = 0.f;
    for (int i = t; i < D; i += 256) { float d = row[i] - mu; vv += d * d; }
    red[t] = vv; __syncthreads();
    for (int k = 128; k > 0; k >>= 1) { if (t < k) red[t] += red[t + k]; __syncthreads(); }
    float inv = rsqrtf(red[0] / D + 1e-5f);
    for (int i = t; i < D; i += 256) row[i] = (row[i] - mu) * inv * w[i] + b[i];
}
// GVP vector norm: v[r,:,:] /= sqrt(mean_nv(sum_c v^2))
__global__ void k_vscale(float* v, int V) {
    __shared__ float red[256];
    float* p = v + (long)blockIdx.x * 3L * V;
    int t = threadIdx.x;
    float s = 0.f;
    for (int i = t; i < 3 * V; i += 256) { float x = p[i]; s += x * x; }
    red[t] = s; __syncthreads();
    for (int k = 128; k > 0; k >>= 1) { if (t < k) red[t] += red[t + k]; __syncthreads(); }
    float f = rsqrtf(fmaxf(red[0] / V, 1e-8f));
    for (int i = t; i < 3 * V; i += 256) p[i] *= f;
}
// edge gather-concat: scalars  out[e] = [s[src], es[e], s[dst]]
__global__ void k_edge_cat_s(const float* s, int ns, const bf16_t* es, int se,
                             const int* src, const int* dst, bf16_t* out, long E) {
    long e = blockIdx.x; if (e >= E) return;
    int a = src[e], b = dst[e];
    int D = 2 * ns + se;
    bf16_t* o = out + e * (long)D;
    const float* sa = s + (long)a * ns;
    const float* sb = s + (long)b * ns;
    const bf16_t* ee = es + e * (long)se;
    for (int i = threadIdx.x; i < ns; i += blockDim.x) o[i] = f2b(sa[i]);
    for (int i = threadIdx.x; i < se; i += blockDim.x) o[ns + i] = ee[i];
    for (int i = threadIdx.x; i < ns; i += blockDim.x) o[ns + se + i] = f2b(sb[i]);
}
// edge gather-concat: vectors ([.,3,V] layout)
__global__ void k_edge_cat_v(const float* v, int nv, const bf16_t* ev, int ve,
                             const int* src, const int* dst, bf16_t* out, long E) {
    long e = blockIdx.x; if (e >= E) return;
    int a = src[e], b = dst[e];
    int D = 2 * nv + ve;
    for (int c = 0; c < 3; ++c) {
        bf16_t* o = out + (e * 3 + c) * (long)D;
        const float* va = v + ((long)a * 3 + c) * nv;
        const float* vb = v + ((long)b * 3 + c) * nv;
        const bf16_t* ee = ev + (e * 3 + c) * (long)ve;
        for (int i = threadIdx.x; i < nv; i += blockDim.x) o[i] = f2b(va[i]);
        for (int i = threadIdx.x; i < ve; i += blockDim.x) o[nv + i] = ee[i];
        for (int i = threadIdx.x; i < nv; i += blockDim.x) o[nv + ve + i] = f2b(vb[i]);
    }
}

// ---------------------------------------------------------------------------
// Host-side orchestration
// ---------------------------------------------------------------------------
static char*  g_ws;
static size_t g_off;
static void*  walloc(size_t bytes) {
    size_t o = (g_off + 255) & ~(size_t)255;
    g_off = o + bytes;
    return g_ws + o;
}
static float*  falloc(long n) { return (float*) walloc((size_t)n * 4); }
static bf16_t* balloc(long n) { return (bf16_t*)walloc((size_t)n * 2); }
static inline dim3 g1d(long n) { return dim3((unsigned)((n + 255) / 256)); }

static void gemm(hipStream_t st, const bf16_t* A, const bf16_t* W, const float* bias,
                 float* C, bf16_t* Cb, long M, int N, int K) {
    dim3 grid((unsigned)((M + BM - 1) / BM), (unsigned)((N + BN - 1) / BN));
    k_gemm<<<grid, 256, 0, st>>>(A, W, bias, C, Cb, (int)M, N, K);
}
static bf16_t* cvt_w(hipStream_t st, const float* w, long n) {
    bf16_t* p = balloc(n);
    k_map<<<g1d(n), 256, 0, st>>>(w, p, n, 0);
    return p;
}

struct GVPp { const float *wh, *ws_b, *ws_w, *wsv_b, *wsv_w, *wv; };
struct LNp  { const float *b, *w; };
struct Convp { GVPp f0, f1; LNp ln0, ln1; GVPp m0, m1, m2; };

static const float* nxt(void* const* din, int& pi) { return (const float*)din[pi++]; }
static GVPp rd_gvp(void* const* din, int& pi, bool vec) {
    GVPp g = {};                              // sorted keys: wh, ws_b, ws_w, (wsv_b, wsv_w, wv)
    g.wh = nxt(din, pi); g.ws_b = nxt(din, pi); g.ws_w = nxt(din, pi);
    if (vec) { g.wsv_b = nxt(din, pi); g.wsv_w = nxt(din, pi); g.wv = nxt(din, pi); }
    return g;
}
static LNp rd_ln(void* const* din, int& pi) {
    LNp l; l.b = nxt(din, pi); l.w = nxt(din, pi); return l;   // sorted: b, w
}
static Convp rd_conv(void* const* din, int& pi) {
    Convp c;                                  // sorted keys: ff[2], ln0, ln1, msg[3]
    c.f0 = rd_gvp(din, pi, true); c.f1 = rd_gvp(din, pi, true);
    c.ln0 = rd_ln(din, pi); c.ln1 = rd_ln(din, pi);
    c.m0 = rd_gvp(din, pi, true); c.m1 = rd_gvp(din, pi, true); c.m2 = rd_gvp(din, pi, true);
    return c;
}

// Geometric Vector Perceptron (vector_gate). s_in bf16 [R,si], v_in bf16 [R,3,vi].
static void gvp_h(hipStream_t st, const GVPp& p, long R,
                  const bf16_t* s_in, const bf16_t* v_in,
                  int si, int vi, int so, int vo, int sact, int vact,
                  float* s_f, bf16_t* s_b, float* v_f, bf16_t* v_b)
{
    const int h = (vi > vo) ? vi : vo;
    const size_t mk = g_off;
    bf16_t* whb = cvt_w(st, p.wh, (long)h * vi);
    float*  vh  = falloc(R * 3L * h);
    bf16_t* vhb = vo ? balloc(R * 3L * h) : nullptr;
    gemm(st, v_in, whb, nullptr, vh, vhb, R * 3, h, vi);          // vh = v @ wh^T
    float* vn = falloc(R * (long)h);
    k_vnorm<<<g1d(R * (long)h), 256, 0, st>>>(vh, vn, R, h);
    bf16_t* cat = balloc(R * (long)(si + h));
    k_cat<bf16_t, float><<<g1d(R * (long)(si + h)), 256, 0, st>>>(s_in, si, vn, h, cat, R);
    bf16_t* wsb = cvt_w(st, p.ws_w, (long)so * (si + h));
    gemm(st, cat, wsb, p.ws_b, s_f, nullptr, R, so, si + h);      // s_out (pre-act)
    if (vo > 0) {
        bf16_t* wvb = cvt_w(st, p.wv, (long)vo * h);
        gemm(st, vhb, wvb, nullptr, v_f, nullptr, R * 3, vo, h);  // v_out = vh @ wv^T
        bf16_t* gin = balloc(R * (long)so);
        k_map<<<g1d(R * (long)so), 256, 0, st>>>(s_f, gin, R * (long)so, vact ? 2 : 0);
        bf16_t* wsvb = cvt_w(st, p.wsv_w, (long)vo * so);
        float* gate = falloc(R * (long)vo);
        gemm(st, gin, wsvb, p.wsv_b, gate, nullptr, R, vo, so);
        k_gate_mul<<<g1d(R * 3L * vo), 256, 0, st>>>(v_f, v_b, gate, R, vo);
    }
    if (s_b) k_map<<<g1d(R * (long)so), 256, 0, st>>>(s_f, s_b, R * (long)so, sact ? 1 : 0);
    if (sact) k_relu_ip<<<g1d(R * (long)so), 256, 0, st>>>(s_f, R * (long)so);
    g_off = mk;   // safe: single stream serializes reuse
}

static void conv_layer(hipStream_t st, const Convp& L,
                       float* s, float* v, const bf16_t* es, int se,
                       const bf16_t* ev, int ve,
                       const int* src, const int* dst, const float* cnt,
                       long Nn, long E)
{
    const size_t mk = g_off;
    const int si0 = 2 * NS + se, vi0 = 2 * NV + ve;
    bf16_t* msin = balloc(E * (long)si0);
    bf16_t* mvin = balloc(E * 3L * vi0);
    k_edge_cat_s<<<(unsigned)E, 256, 0, st>>>(s, NS, es, se, src, dst, msin, E);
    k_edge_cat_v<<<(unsigned)E, 256, 0, st>>>(v, NV, ev, ve, src, dst, mvin, E);

    float*  tmps = falloc(E * (long)NS);
    float*  tmpv = falloc(E * 3L * NV);
    bf16_t* s1 = balloc(E * (long)NS); bf16_t* v1 = balloc(E * 3L * NV);
    gvp_h(st, L.m0, E, msin, mvin, si0, vi0, NS, NV, 1, 1, tmps, s1, tmpv, v1);
    bf16_t* s2 = balloc(E * (long)NS); bf16_t* v2 = balloc(E * 3L * NV);
    gvp_h(st, L.m1, E, s1, v1, NS, NV, NS, NV, 1, 1, tmps, s2, tmpv, v2);
    float* ms = falloc(E * (long)NS);
    float* mv = falloc(E * 3L * NV);
    gvp_h(st, L.m2, E, s2, v2, NS, NV, NS, NV, 0, 0, ms, nullptr, mv, nullptr);

    float* accS = falloc(Nn * (long)NS);
    float* accV = falloc(Nn * 3L * NV);
    hipMemsetAsync(accS, 0, (size_t)Nn * NS * 4, st);
    hipMemsetAsync(accV, 0, (size_t)Nn * 3 * NV * 4, st);
    k_scatter_add<<<g1d(E * (long)NS), 256, 0, st>>>(ms, dst, accS, E, NS);
    k_scatter_add<<<g1d(E * 3L * NV), 256, 0, st>>>(mv, dst, accV, E, 3 * NV);
    k_mean_res<<<g1d(Nn * (long)NS), 256, 0, st>>>(s, accS, cnt, Nn, NS);
    k_mean_res<<<g1d(Nn * 3L * NV), 256, 0, st>>>(v, accV, cnt, Nn, 3 * NV);
    k_ln<<<(unsigned)Nn, 256, 0, st>>>(s, L.ln0.w, L.ln0.b, NS);
    k_vscale<<<(unsigned)Nn, 256, 0, st>>>(v, NV);

    bf16_t* sb = balloc(Nn * (long)NS);
    bf16_t* vb = balloc(Nn * 3L * NV);
    k_map<<<g1d(Nn * (long)NS), 256, 0, st>>>(s, sb, Nn * (long)NS, 0);
    k_map<<<g1d(Nn * 3L * NV), 256, 0, st>>>(v, vb, Nn * 3L * NV, 0);
    float*  f0s = falloc(Nn * 4L * NS); bf16_t* f0sb = balloc(Nn * 4L * NS);
    float*  f0v = falloc(Nn * 6L * NV); bf16_t* f0vb = balloc(Nn * 6L * NV);
    gvp_h(st, L.f0, Nn, sb, vb, NS, NV, 4 * NS, 2 * NV, 1, 1, f0s, f0sb, f0v, f0vb);
    float* fs = falloc(Nn * (long)NS);
    float* fv = falloc(Nn * 3L * NV);
    gvp_h(st, L.f1, Nn, f0sb, f0vb, 4 * NS, 2 * NV, NS, NV, 0, 0, fs, nullptr, fv, nullptr);
    k_add_ip<<<g1d(Nn * (long)NS), 256, 0, st>>>(s, fs, Nn * (long)NS);
    k_add_ip<<<g1d(Nn * 3L * NV), 256, 0, st>>>(v, fv, Nn * 3L * NV);
    k_ln<<<(unsigned)Nn, 256, 0, st>>>(s, L.ln1.w, L.ln1.b, NS);
    k_vscale<<<(unsigned)Nn, 256, 0, st>>>(v, NV);
    g_off = mk;
}

extern "C" void kernel_launch(void* const* d_in, const int* in_sizes, int n_in,
                              void* d_out, int out_size, void* d_ws, size_t ws_size,
                              hipStream_t stream)
{
    g_ws = (char*)d_ws; g_off = 0;
    const float* hVs = (const float*)d_in[0];
    const float* hVv = (const float*)d_in[1];
    const int*   ei  = (const int*)d_in[2];
    const float* hEs = (const float*)d_in[3];
    const float* hEv = (const float*)d_in[4];
    const float* msa = (const float*)d_in[5];
    const int*   seq = (const int*)d_in[6];
    const long N = in_sizes[0] / 6;
    const long E = in_sizes[2] / 2;
    const int* src = ei;
    const int* dst = ei + E;

    // params in JAX pytree (sorted dict key) order, leaves starting at d_in[7]
    int pi = 7;
    const float* W_M_b = nxt(d_in, pi);
    const float* W_M_w = nxt(d_in, pi);
    const float* W_S   = nxt(d_in, pi);
    GVPp  We    = rd_gvp(d_in, pi, true);
    LNp   We_ln = rd_ln(d_in, pi);
    GVPp  Wout  = rd_gvp(d_in, pi, false);
    GVPp  Wv    = rd_gvp(d_in, pi, true);
    LNp   Wv_ln = rd_ln(d_in, pi);
    Convp dec[4]; for (int i = 0; i < 4; ++i) dec[i] = rd_conv(d_in, pi);
    const float* di_b1 = nxt(d_in, pi);
    const float* di_b2 = nxt(d_in, pi);
    LNp di_ln1 = rd_ln(d_in, pi);
    LNp di_ln2 = rd_ln(d_in, pi);
    const float* di_w1 = nxt(d_in, pi);
    const float* di_w2 = nxt(d_in, pi);
    Convp enc[4]; for (int i = 0; i < 4; ++i) enc[i] = rd_conv(d_in, pi);

    // persistent state
    float* cnt = falloc(N);
    hipMemsetAsync(cnt, 0, (size_t)N * 4, stream);
    k_count<<<g1d(E), 256, 0, stream>>>(dst, cnt, E);
    float*  s   = falloc(N * (long)NS);
    float*  v   = falloc(N * 3L * NV);
    bf16_t* esb = balloc(E * (long)ES);
    bf16_t* evb = balloc(E * 3L * EV);
    bf16_t* es2 = balloc(E * (long)(ES + NAA));
    const size_t base = g_off;

    { // node embedding: W_v GVP (no acts) + GVP layernorm
        bf16_t* s6 = balloc(N * 6);
        k_map<<<g1d(N * 6), 256, 0, stream>>>(hVs, s6, N * 6, 0);
        bf16_t* v3 = balloc(N * 9);
        k_tr_v<<<g1d(N * 9), 256, 0, stream>>>(hVv, v3, N, 3);
        gvp_h(stream, Wv, N, s6, v3, 6, 3, NS, NV, 0, 0, s, nullptr, v, nullptr);
        k_ln<<<(unsigned)N, 256, 0, stream>>>(s, Wv_ln.w, Wv_ln.b, NS);
        k_vscale<<<(unsigned)N, 256, 0, stream>>>(v, NV);
        g_off = base;
    }
    { // edge embedding: W_e GVP + layernorm -> bf16
        bf16_t* e32 = balloc(E * 32);
        k_map<<<g1d(E * 32), 256, 0, stream>>>(hEs, e32, E * 32, 0);
        bf16_t* ev1 = balloc(E * 3);
        k_tr_v<<<g1d(E * 3), 256, 0, stream>>>(hEv, ev1, E, 1);
        float* esf = falloc(E * (long)ES);
        float* evf = falloc(E * 3L * EV);
        gvp_h(stream, We, E, e32, ev1, 32, 1, ES, EV, 0, 0, esf, nullptr, evf, nullptr);
        k_ln<<<(unsigned)E, 256, 0, stream>>>(esf, We_ln.w, We_ln.b, ES);
        k_vscale<<<(unsigned)E, 256, 0, stream>>>(evf, EV);
        k_map<<<g1d(E * (long)ES), 256, 0, stream>>>(esf, esb, E * (long)ES, 0);
        k_map<<<g1d(E * 3L * EV), 256, 0, stream>>>(evf, evb, E * 3L * EV, 0);
        g_off = base;
    }

    for (int l = 0; l < 4; ++l)
        conv_layer(stream, enc[l], s, v, esb, ES, evb, EV, src, dst, cnt, N, E);

    { // es = cat(es, W_S[seq][src])
        bf16_t* hs = balloc(E * (long)NAA);
        k_gather_hs<<<g1d(E * (long)NAA), 256, 0, stream>>>(W_S, seq, src, hs, E, NAA);
        k_cat<bf16_t, bf16_t><<<g1d(E * (long)(ES + NAA)), 256, 0, stream>>>(esb, ES, hs, NAA, es2, E);
        g_off = base;
    }
    { // decoder input MLP
        bf16_t* msab = balloc(N * (long)MSAD);
        k_map<<<g1d(N * (long)MSAD), 256, 0, stream>>>(msa, msab, N * (long)MSAD, 0);
        bf16_t* wmb = cvt_w(stream, W_M_w, (long)NS * MSAD);
        float* hM = falloc(N * (long)NS);
        gemm(stream, msab, wmb, W_M_b, hM, nullptr, N, NS, MSAD);
        bf16_t* x = balloc(N * 2L * NS);
        k_cat<float, float><<<g1d(N * 2L * NS), 256, 0, stream>>>(s, NS, hM, NS, x, N);
        bf16_t* w1b = cvt_w(stream, di_w1, 2L * NS * 2L * NS);
        float* x1 = falloc(N * 2L * NS);
        gemm(stream, x, w1b, di_b1, x1, nullptr, N, 2 * NS, 2 * NS);
        k_ln<<<(unsigned)N, 256, 0, stream>>>(x1, di_ln1.w, di_ln1.b, 2 * NS);
        k_relu_ip<<<g1d(N * 2L * NS), 256, 0, stream>>>(x1, N * 2L * NS);
        bf16_t* x1b = balloc(N * 2L * NS);
        k_map<<<g1d(N * 2L * NS), 256, 0, stream>>>(x1, x1b, N * 2L * NS, 0);
        bf16_t* w2b = cvt_w(stream, di_w2, (long)NS * 2L * NS);
        float* x2 = falloc(N * (long)NS);
        gemm(stream, x1b, w2b, di_b2, x2, nullptr, N, NS, 2 * NS);
        k_ln<<<(unsigned)N, 256, 0, stream>>>(x2, di_ln2.w, di_ln2.b, NS);
        k_copy_f<<<g1d(N * (long)NS), 256, 0, stream>>>(s, x2, N * (long)NS);
        g_off = base;
    }

    for (int l = 0; l < 4; ++l)
        conv_layer(stream, dec[l], s, v, es2, ES + NAA, evb, EV, src, dst, cnt, N, E);

    { // output GVP (vo=0) -> logits [N,20] f32 straight into d_out
        bf16_t* sb = balloc(N * (long)NS);
        bf16_t* vb = balloc(N * 3L * NV);
        k_map<<<g1d(N * (long)NS), 256, 0, stream>>>(s, sb, N * (long)NS, 0);
        k_map<<<g1d(N * 3L * NV), 256, 0, stream>>>(v, vb, N * 3L * NV, 0);
        gvp_h(stream, Wout, N, sb, vb, NS, NV, 20, 0, 0, 0,
              (float*)d_out, nullptr, nullptr, nullptr);
    }
}
